// DecoderSelfAttention_89180700934478
// MI455X (gfx1250) — matrix operand order repaired and checked by reading, not compiled, not run on hardware
//
#include <hip/hip_runtime.h>
#include <hip/hip_bf16.h>

// ---------------- CDNA5 WMMA / vector types ----------------
typedef __attribute__((ext_vector_type(16))) _Float16 v16h;
typedef __attribute__((ext_vector_type(8)))  _Float16 v8h;
typedef __attribute__((ext_vector_type(4)))  _Float16 v4h;
typedef __attribute__((ext_vector_type(8)))  float    v8f;
typedef __attribute__((ext_vector_type(4)))  unsigned int v4u;
typedef __attribute__((ext_vector_type(4)))  int      v4i;
typedef __attribute__((ext_vector_type(8)))  int      v8i;

#define WMMA_F16(A, B, C) \
  __builtin_amdgcn_wmma_f32_16x16x32_f16(false, (A), false, (B), (short)0, (C), false, false)

union AF { v16h v; v8h h[2]; };

static __device__ __forceinline__ v8h ld8(const _Float16* p) { return *(const v8h*)p; }

// ---- async global->LDS copy (ASYNCcnt path), per-lane 16B ----
static __device__ __forceinline__ void async_b128(void* lds_ptr, const void* gptr) {
  // LDS byte offset = low 32 bits of the generic address (ISA 10.2 aperture rule)
  unsigned int l = (unsigned int)(size_t)lds_ptr;
  asm volatile("global_load_async_to_lds_b128 %0, %1, off"
               :: "v"(l), "v"(gptr) : "memory");
}
static __device__ __forceinline__ void wait_async0() {
  asm volatile("s_wait_asynccnt 0x0" ::: "memory");
}

// ---- Tensor Data Mover: 2D tile load, f16 elements (D# per ISA 08 §8.3/8.4) ----
static __device__ __forceinline__ void tdm_load_2d_f16(
    unsigned int lds_off, const void* gaddr,
    unsigned int tensor_d0, unsigned int tensor_d1, unsigned int stride_d0,
    unsigned int tile_d0, unsigned int tile_d1,
    unsigned int pad_interval_code, unsigned int pad_amount_code, bool pad_en) {
  unsigned long long ga = (unsigned long long)(size_t)gaddr;
  v4u g0;
  g0[0] = 1u;                                     // count=1, user descriptor
  g0[1] = lds_off;                                // lds_addr (bytes)
  g0[2] = (unsigned int)ga;                       // global_addr lo
  g0[3] = (unsigned int)(ga >> 32) | 0x80000000u; // addr hi | type=2 (bits 127:126)
  v8i g1;
  unsigned int w0 = (1u << 16);                   // data_size=1 -> 2 bytes; wg_mask=0
  if (pad_en) w0 |= (1u << 20) | (pad_interval_code << 22) | (pad_amount_code << 25);
  g1[0] = (int)w0;
  g1[1] = (int)((tensor_d0 & 0xffffu) << 16);                                   // dim0[15:0]
  g1[2] = (int)(((tensor_d0 >> 16) & 0xffffu) | ((tensor_d1 & 0xffffu) << 16)); // dim0 hi | dim1 lo
  g1[3] = (int)(((tensor_d1 >> 16) & 0xffffu) | ((tile_d0 & 0xffffu) << 16));   // dim1 hi | tile0
  g1[4] = (int)(tile_d1 & 0xffffu);               // tile1 | tile2=0
  g1[5] = (int)stride_d0;                         // dim0 stride lo32
  g1[6] = 0;                                      // stride0 hi | stride1 lo
  g1[7] = 0;
  v4i gz = {0, 0, 0, 0};
#if __clang_major__ >= 23
  v8i gz8 = {0, 0, 0, 0, 0, 0, 0, 0};
  __builtin_amdgcn_tensor_load_to_lds(g0, g1, gz, gz, gz8, 0);
#else
  __builtin_amdgcn_tensor_load_to_lds(g0, g1, gz, gz, 0);
#endif
}

// ---------------- fp32 -> fp16 conversion ----------------
__global__ void cvt_f32_f16(const float* __restrict__ s, _Float16* __restrict__ d, int n4) {
  int i = blockIdx.x * blockDim.x + threadIdx.x;
  if (i < n4) {
    float4 x = ((const float4*)s)[i];
    v4h y;
    y[0] = (_Float16)x.x; y[1] = (_Float16)x.y;
    y[2] = (_Float16)x.z; y[3] = (_Float16)x.w;
    ((v4h*)d)[i] = y;
  }
}

// ---------------- f16 x f16 -> f32-accum GEMM (double-buffered, async A) ----------------
// C(MxN) = A(MxK,row-major) * B(KxN,row-major). Output f16 (Ch) or f32 (Cf).
// Block: 256 thr = 8 waves; tile 128x128; wave tile 32x64 (2x4 WMMA tiles).
#define TM 128
#define TN 128
#define KC 32
#define LDK 40   // padded LDS row stride (f16): 80B => conflict-free b128 frag reads

__global__ void __launch_bounds__(256)
gemm_f16(const _Float16* __restrict__ A, const _Float16* __restrict__ B,
         _Float16* __restrict__ Ch, float* __restrict__ Cf,
         int M, int N, int K) {
  __shared__ _Float16 lA[2][TM][LDK];
  __shared__ _Float16 lB[2][TN][LDK];

  const int tid  = threadIdx.x;
  const int lane = tid & 31;
  const int wave = tid >> 5;
  const int half = lane >> 4;
  const int l16  = lane & 15;
  const int wm   = wave >> 1;  // 0..3 (M)
  const int wn   = wave & 1;   // 0..1 (N)
  const int m0   = blockIdx.y * TM;
  const int n0   = blockIdx.x * TN;

  // per-thread staging coordinates (2 passes of 256 threads)
  const int arow[2] = { (tid + 0) >> 2, (tid + 256) >> 2 };
  const int acol    = (tid & 3) * 8;
  const int bk[2]   = { (tid + 0) >> 4, (tid + 256) >> 4 };
  const int bn8     = (tid & 15) * 8;

  const v8f vzero = {0.f,0.f,0.f,0.f,0.f,0.f,0.f,0.f};
  v8f acc[2][4];
#pragma unroll
  for (int mi = 0; mi < 2; mi++)
#pragma unroll
    for (int ni = 0; ni < 4; ni++) acc[mi][ni] = vzero;

  // ---- prologue: stage tile k0=0 into buffer 0 ----
#pragma unroll
  for (int p = 0; p < 2; p++)
    async_b128(&lA[0][arow[p]][acol], &A[(size_t)(m0 + arow[p]) * K + acol]);
  {
    v8h b0[2];
#pragma unroll
    for (int p = 0; p < 2; p++)
      b0[p] = ld8(&B[(size_t)bk[p] * N + n0 + bn8]);
#pragma unroll
    for (int p = 0; p < 2; p++)
#pragma unroll
      for (int i = 0; i < 8; i++) lB[0][bn8 + i][bk[p]] = b0[p][i];
  }
  wait_async0();
  __syncthreads();

  int cur = 0;
  for (int k0 = 0; k0 < K; k0 += KC) {
    const int nxt = cur ^ 1;
    const bool hn = (k0 + KC) < K;

    // ---- issue next tile's staging before computing current ----
    v8h bnreg[2];
    if (hn) {
#pragma unroll
      for (int p = 0; p < 2; p++)
        async_b128(&lA[nxt][arow[p]][acol],
                   &A[(size_t)(m0 + arow[p]) * K + (k0 + KC) + acol]);
#pragma unroll
      for (int p = 0; p < 2; p++)
        bnreg[p] = ld8(&B[(size_t)(k0 + KC + bk[p]) * N + n0 + bn8]);
    }

    // ---- compute current tile: load ALL fragments first, then 8 WMMAs ----
    // A frags (16x32, ISA 7.12.2): VGPR0-3 K=8h..8h+7, VGPR4-7 K=16+8h..+7
    AF af[2], bf[4];
#pragma unroll
    for (int mi = 0; mi < 2; mi++) {
      int row = wm * 32 + mi * 16 + l16;
      af[mi].h[0] = *(const v8h*)&lA[cur][row][8 * half];
      af[mi].h[1] = *(const v8h*)&lA[cur][row][16 + 8 * half];
    }
    // B frags (32x16): N = lane&15, VGPR0-3 K=8h..8h+7, VGPR4-7 K=16+8h..+7
#pragma unroll
    for (int ni = 0; ni < 4; ni++) {
      int n = wn * 64 + ni * 16 + l16;
      bf[ni].h[0] = *(const v8h*)&lB[cur][n][8 * half];
      bf[ni].h[1] = *(const v8h*)&lB[cur][n][16 + 8 * half];
    }
#pragma unroll
    for (int ni = 0; ni < 4; ni++)
#pragma unroll
      for (int mi = 0; mi < 2; mi++)
        acc[mi][ni] = WMMA_F16(af[mi].v, bf[ni].v, acc[mi][ni]);

    // ---- retire next tile's B stores (global loads have landed by now) ----
    if (hn) {
#pragma unroll
      for (int p = 0; p < 2; p++)
#pragma unroll
        for (int i = 0; i < 8; i++) lB[nxt][bn8 + i][bk[p]] = bnreg[p][i];
    }
    wait_async0();
    __syncthreads();
    cur = nxt;
  }

  // epilogue (C layout: row M = vgpr + 8*half, col N = lane&15)
  if (Cf) {
#pragma unroll
    for (int mi = 0; mi < 2; mi++)
#pragma unroll
      for (int ni = 0; ni < 4; ni++) {
        int colg = n0 + wn * 64 + ni * 16 + l16;
#pragma unroll
        for (int r = 0; r < 8; r++) {
          int rowg = m0 + wm * 32 + mi * 16 + 8 * half + r;
          Cf[(size_t)rowg * N + colg] = acc[mi][ni][r];
        }
      }
  } else {
#pragma unroll
    for (int mi = 0; mi < 2; mi++)
#pragma unroll
      for (int ni = 0; ni < 4; ni++) {
        int colg = n0 + wn * 64 + ni * 16 + l16;
#pragma unroll
        for (int r = 0; r < 8; r++) {
          int rowg = m0 + wm * 32 + mi * 16 + 8 * half + r;
          Ch[(size_t)rowg * N + colg] = (_Float16)acc[mi][ni][r];
        }
      }
  }
}

// ---------------- flash attention (causal + pad mask) ----------------
// grid (qblock=32, head=16, batch=2); block 128 thr = 4 waves.
// K tile staged by the Tensor Data Mover (pad_enable reproduces the 72-hw stride).
#define L_SEQ 2048
#define DMODEL 1024
#define DH 64

__global__ void __launch_bounds__(128)
attn_fwd(const _Float16* __restrict__ Qh, const _Float16* __restrict__ Kh,
         const _Float16* __restrict__ Vh, const float* __restrict__ am,
         _Float16* __restrict__ Ctx) {
  __shared__ _Float16 lK[64][72];      // keys row-major [key][d]   (TDM destination)
  __shared__ _Float16 lV[64][72];      // V transposed [d][key]
  __shared__ _Float16 lP[4][16][72];   // per-wave P tile [row][key]

  const int tid  = threadIdx.x;
  const int lane = tid & 31;
  const int wave = tid >> 5;
  const int half = lane >> 4;
  const int l16  = lane & 15;
  const int qb = blockIdx.x, h = blockIdx.y, b = blockIdx.z;
  const int fo = h * DH;
  const size_t base = (size_t)b * L_SEQ;
  const float NEG = -__builtin_inff();

  // Q fragments (A layout), rows = qb*64 + wave*16 + (lane&15)
  const int qrow = qb * 64 + wave * 16 + l16;
  AF qf[2];
#pragma unroll
  for (int ks = 0; ks < 2; ks++) {
    const _Float16* qp = &Qh[(base + qrow) * DMODEL + fo + ks * 32];
    qf[ks].h[0] = ld8(qp + 8 * half);
    qf[ks].h[1] = ld8(qp + 16 + 8 * half);
  }

  float mrow[8], lrow[8], amq[8];
  const v8f vzero = {0.f,0.f,0.f,0.f,0.f,0.f,0.f,0.f};
  v8f o[4];
#pragma unroll
  for (int r = 0; r < 8; r++) {
    mrow[r] = NEG; lrow[r] = 0.f;
    amq[r] = am[base + qb * 64 + wave * 16 + 8 * half + r];
  }
#pragma unroll
  for (int jd = 0; jd < 4; jd++) o[jd] = vzero;

  const unsigned int lK_off = (unsigned int)(size_t)&lK[0][0];

  for (int kb = 0; kb <= qb; kb++) {
    __syncthreads();
    // ---- K tile via TDM: 64x64 f16, dim0 stride 1024, pad 16B per 128B row ----
    if (wave == 0) {
      tdm_load_2d_f16(lK_off, &Kh[(base + kb * 64) * DMODEL + fo],
                      /*tensor_d0=*/DMODEL, /*tensor_d1=*/2 * L_SEQ,
                      /*stride_d0=*/DMODEL,
                      /*tile_d0=*/64, /*tile_d1=*/64,
                      /*pad_interval(128B)=*/4, /*pad_amount(16B)=*/3, true);
    }
    // ---- stage V tile transposed (needs element shuffle: sync path) ----
#pragma unroll
    for (int p = 0; p < 4; p++) {
      int c = tid + p * 128;
      int key = c >> 3, d8 = (c & 7) * 8;
      v8h g = ld8(&Vh[(base + kb * 64 + key) * DMODEL + fo + d8]);
#pragma unroll
      for (int i = 0; i < 8; i++) lV[d8 + i][key] = g[i];
    }
    if (wave == 0) __builtin_amdgcn_s_wait_tensorcnt(0);
    __syncthreads();

    // S = Q K^T : 4 key tiles of 16, contraction over d=64 (2 k-steps)
    // load all 8 B-fragments first so the ds_loads pipeline, then 8 WMMAs
    AF kf[4][2];
#pragma unroll
    for (int jn = 0; jn < 4; jn++) {
      int n = jn * 16 + l16;
#pragma unroll
      for (int ks = 0; ks < 2; ks++) {
        kf[jn][ks].h[0] = *(const v8h*)&lK[n][ks * 32 + 8 * half];
        kf[jn][ks].h[1] = *(const v8h*)&lK[n][ks * 32 + 16 + 8 * half];
      }
    }
    v8f st[4];
#pragma unroll
    for (int jn = 0; jn < 4; jn++) {
      v8f s = vzero;
#pragma unroll
      for (int ks = 0; ks < 2; ks++)
        s = WMMA_F16(qf[ks].v, kf[jn][ks].v, s);
      st[jn] = s;
    }

    // scale (1/sqrt(1024)=1/32) + causal & pad masks
#pragma unroll
    for (int jn = 0; jn < 4; jn++) {
      int ktok = kb * 64 + jn * 16 + l16;
      float amk = am[base + ktok];
#pragma unroll
      for (int r = 0; r < 8; r++) {
        int qtok = qb * 64 + wave * 16 + 8 * half + r;
        float sv = st[jn][r] * 0.03125f;
        bool masked = (ktok > qtok) || (amq[r] * amk < 0.5f);
        st[jn][r] = masked ? NEG : sv;
      }
    }

    // online softmax: reduce across the 16 lanes holding one row's 16 cols
#pragma unroll
    for (int r = 0; r < 8; r++) {
      float v = fmaxf(fmaxf(st[0][r], st[1][r]), fmaxf(st[2][r], st[3][r]));
      v = fmaxf(v, __shfl_xor(v, 1, 32));
      v = fmaxf(v, __shfl_xor(v, 2, 32));
      v = fmaxf(v, __shfl_xor(v, 4, 32));
      v = fmaxf(v, __shfl_xor(v, 8, 32));
      float mnew = fmaxf(mrow[r], v);
      float a = (mnew == mrow[r]) ? 1.0f : __expf(mrow[r] - mnew);
      mrow[r] = mnew;
      float psum = 0.f;
#pragma unroll
      for (int jn = 0; jn < 4; jn++) {
        float sv = st[jn][r];
        float p = (sv == NEG) ? 0.f : __expf(sv - mnew);
        st[jn][r] = p;
        psum += p;
      }
      psum += __shfl_xor(psum, 1, 32);
      psum += __shfl_xor(psum, 2, 32);
      psum += __shfl_xor(psum, 4, 32);
      psum += __shfl_xor(psum, 8, 32);
      lrow[r] = lrow[r] * a + psum;
#pragma unroll
      for (int jd = 0; jd < 4; jd++) o[jd][r] *= a;
      // C-layout -> LDS so we can re-read in A-fragment layout
#pragma unroll
      for (int jn = 0; jn < 4; jn++)
        lP[wave][8 * half + r][jn * 16 + l16] = (_Float16)st[jn][r];
    }
    __syncthreads();

    // O += P @ V : contraction over 64 keys (2 k-steps), 4 d-tiles
    // load all fragments first, then 8 WMMAs
    AF pf[2], vf[4][2];
#pragma unroll
    for (int ks = 0; ks < 2; ks++) {
      pf[ks].h[0] = *(const v8h*)&lP[wave][l16][ks * 32 + 8 * half];
      pf[ks].h[1] = *(const v8h*)&lP[wave][l16][ks * 32 + 16 + 8 * half];
    }
#pragma unroll
    for (int jd = 0; jd < 4; jd++) {
      int d = jd * 16 + l16;
#pragma unroll
      for (int ks = 0; ks < 2; ks++) {
        vf[jd][ks].h[0] = *(const v8h*)&lV[d][ks * 32 + 8 * half];
        vf[jd][ks].h[1] = *(const v8h*)&lV[d][ks * 32 + 16 + 8 * half];
      }
    }
#pragma unroll
    for (int jd = 0; jd < 4; jd++)
#pragma unroll
      for (int ks = 0; ks < 2; ks++)
        o[jd] = WMMA_F16(pf[ks].v, vf[jd][ks].v, o[jd]);
  }

  // normalize and write ctx (b, l, h*64 + d)
#pragma unroll
  for (int r = 0; r < 8; r++) {
    float inv = 1.0f / lrow[r];
    int tok = qb * 64 + wave * 16 + 8 * half + r;
#pragma unroll
    for (int jd = 0; jd < 4; jd++)
      Ctx[(base + tok) * DMODEL + fo + jd * 16 + l16] = (_Float16)(o[jd][r] * inv);
  }
}

// ---------------- launcher ----------------
extern "C" void kernel_launch(void* const* d_in, const int* in_sizes, int n_in,
                              void* d_out, int out_size, void* d_ws, size_t ws_size,
                              hipStream_t stream) {
  const float* X  = (const float*)d_in[0];
  const float* am = (const float*)d_in[1];
  const float* wq = (const float*)d_in[2];
  const float* wk = (const float*)d_in[3];
  const float* wv = (const float*)d_in[4];
  const float* wo = (const float*)d_in[5];
  float* out = (float*)d_out;

  const int Mt = 2 * 2048;   // tokens
  const int D  = 1024;

  // workspace layout (f16), ~48 MiB total
  char* w = (char*)d_ws;
  _Float16* Xh  = (_Float16*)(w + (size_t)0);
  _Float16* Wqh = (_Float16*)(w + ((size_t)8  << 20));
  _Float16* Wkh = (_Float16*)(w + ((size_t)10 << 20));
  _Float16* Wvh = (_Float16*)(w + ((size_t)12 << 20));
  _Float16* Woh = (_Float16*)(w + ((size_t)14 << 20));
  _Float16* Qh  = (_Float16*)(w + ((size_t)16 << 20));
  _Float16* Kh  = (_Float16*)(w + ((size_t)24 << 20));
  _Float16* Vh  = (_Float16*)(w + ((size_t)32 << 20));
  _Float16* Ch  = (_Float16*)(w + ((size_t)40 << 20));

  int nx4 = (Mt * D) / 4;
  cvt_f32_f16<<<(nx4 + 255) / 256, 256, 0, stream>>>(X, Xh, nx4);
  int nw4 = (D * D) / 4;
  cvt_f32_f16<<<(nw4 + 255) / 256, 256, 0, stream>>>(wq, Wqh, nw4);
  cvt_f32_f16<<<(nw4 + 255) / 256, 256, 0, stream>>>(wk, Wkh, nw4);
  cvt_f32_f16<<<(nw4 + 255) / 256, 256, 0, stream>>>(wv, Wvh, nw4);
  cvt_f32_f16<<<(nw4 + 255) / 256, 256, 0, stream>>>(wo, Woh, nw4);

  dim3 gg(D / TN, Mt / TM);  // (8, 32)
  gemm_f16<<<gg, 256, 0, stream>>>(Xh, Wqh, Qh, nullptr, Mt, D, D);
  gemm_f16<<<gg, 256, 0, stream>>>(Xh, Wkh, Kh, nullptr, Mt, D, D);
  gemm_f16<<<gg, 256, 0, stream>>>(Xh, Wvh, Vh, nullptr, Mt, D, D);

  dim3 ga(32, 16, 2);        // (qblocks, heads, batch)
  attn_fwd<<<ga, 128, 0, stream>>>(Qh, Kh, Vh, am, Ch);

  gemm_f16<<<gg, 256, 0, stream>>>(Ch, Woh, nullptr, out, Mt, D, D);
}
